// CommAttention_25649544692349
// MI455X (gfx1250) — matrix operand order repaired and checked by reading, not compiled, not run on hardware
//
#include <hip/hip_runtime.h>

// ---------------------------------------------------------------------------
// CommAttention for MI455X (gfx1250): grouped GEMMs + per-head attention,
// all matmuls on v_wmma_f32_16x16x32_bf16 (f32 accumulate).
// Round 2: software-pipelined GEMM K-loop (global loads overlap WMMA),
//          ds_swizzle-based softmax reductions, global_prefetch on W stream.
// ---------------------------------------------------------------------------

typedef __attribute__((ext_vector_type(16))) __bf16 v16bf;
typedef __attribute__((ext_vector_type(8)))  float  v8f;

union Frag16 {
    v16bf v;
    unsigned int w[8];   // 8 dwords = 16 bf16 per lane
};

// round-to-nearest-even float -> bf16 (bit pattern as ushort)
__device__ __forceinline__ unsigned short f2bf(float f) {
    unsigned int x = __builtin_bit_cast(unsigned int, f);
    unsigned int r = x + 0x7FFFu + ((x >> 16) & 1u);
    return (unsigned short)(r >> 16);
}
__device__ __forceinline__ unsigned int pack2bf(float a, float b) {
    return (unsigned int)f2bf(a) | ((unsigned int)f2bf(b) << 16);
}

// single-instruction XOR lane shuffle within the wave (ds_swizzle_b32):
// new_lane = ((lane & 0x1f) | 0) ^ XORM  -- XORM in {1,2,4,8} stays inside
// each 16-lane half, which is exactly the softmax reduction group.
template <int XORM>
__device__ __forceinline__ float swz_xor(float x) {
    int i = __builtin_bit_cast(int, x);
    int r = __builtin_amdgcn_ds_swizzle(i, (XORM << 10) | 0x1f);
    return __builtin_bit_cast(float, r);
}

// ---------------------------------------------------------------------------
// f32 -> bf16 conversion (for h only; weights are converted in-GEMM)
// ---------------------------------------------------------------------------
__global__ __launch_bounds__(256) void cvt_f32_bf16(const float* __restrict__ in,
                                                    unsigned short* __restrict__ out,
                                                    int n) {
    int i = blockIdx.x * 256 + threadIdx.x;
    if (i < n) out[i] = f2bf(in[i]);
}

// ---------------------------------------------------------------------------
// Grouped GEMM:  Out[b, g, :] = A[b, g, :] (bf16, DIN)  x  W[g] (f32 DINxDOUT)
// Block tile: 64(M=b) x 128(N), K-step 32.  8 waves -> 2x4 wave grid, each
// wave owns a 32x32 tile = 2x2 WMMA tiles.  W converted f32->bf16 into LDS.
// Software pipelined: next K-step staged in registers during current WMMAs.
// ---------------------------------------------------------------------------
template <int DIN, int DOUT, bool OUT_F32>
__global__ __launch_bounds__(256) void grouped_gemm_bf16(
    const unsigned short* __restrict__ A,   // (256, 16, DIN) bf16
    const float* __restrict__ W,            // (16, DIN, DOUT) f32
    void* __restrict__ Out)                 // (256, 16, DOUT) bf16 or f32
{
    constexpr int BM = 64, BN = 128, BK = 32;
    constexpr int NSTEP = DIN / BK;
    __shared__ __align__(16) unsigned short As[BM * BK];   // [m][k]
    __shared__ __align__(16) unsigned short Bs[BN * BK];   // [n][k] (transposed)

    const int g  = blockIdx.z;
    const int m0 = blockIdx.y * BM;
    const int n0 = blockIdx.x * BN;
    const int t    = threadIdx.x;
    const int lane = t & 31;
    const int wv   = t >> 5;           // wave 0..7
    const int wm   = (wv & 1) * 32;    // wave row offset in tile
    const int wn   = (wv >> 1) * 32;   // wave col offset in tile
    const int lm   = lane & 15;
    const int half = lane >> 4;

    const float* Wg = W + (size_t)g * DIN * DOUT;

    v8f acc[2][2] = {};

    // A-load coords: one 16B (8 x bf16) load per thread per K-step
    const int am = t >> 2;            // 0..63
    const int ak = (t & 3) * 8;       // 0,8,16,24
    const unsigned short* Arow = A + (size_t)((m0 + am) * 16 + g) * DIN + ak;
    // B-load coords: (n, k-pair), 8 iterations of 256 threads per K-step
    const int bn   = t & 127;
    const int bkp0 = t >> 7;          // 0..1

    // ---- prologue: stage K-step 0 ----
    uint4 av = *reinterpret_cast<const uint4*>(Arow);
    float w0r[8], w1r[8];
#pragma unroll
    for (int i = 0; i < 8; ++i) {
        int kk = (bkp0 + i * 2) * 2;
        w0r[i] = Wg[(size_t)kk * DOUT + n0 + bn];
        w1r[i] = Wg[(size_t)(kk + 1) * DOUT + n0 + bn];
    }
    *reinterpret_cast<uint4*>(&As[am * BK + ak]) = av;
#pragma unroll
    for (int i = 0; i < 8; ++i)
        *reinterpret_cast<unsigned int*>(&Bs[bn * BK + (bkp0 + i * 2) * 2]) =
            pack2bf(w0r[i], w1r[i]);
    __syncthreads();

    for (int step = 0; step < NSTEP; ++step) {
        const int k_next = (step + 1) * BK;
        const bool has_next = (k_next < DIN);

        // ---- issue next K-step's global loads (overlap with WMMA below) ----
        if (has_next) {
            av = *reinterpret_cast<const uint4*>(Arow + k_next);
#pragma unroll
            for (int i = 0; i < 8; ++i) {
                int kk = k_next + (bkp0 + i * 2) * 2;
                w0r[i] = Wg[(size_t)kk * DOUT + n0 + bn];
                w1r[i] = Wg[(size_t)(kk + 1) * DOUT + n0 + bn];
            }
            // keep L2 primed two steps ahead on the big W stream
            if (k_next + BK < DIN)
                __builtin_prefetch(&Wg[(size_t)(k_next + BK) * DOUT + n0 + bn], 0, 0);
        }

        // ---- fragments per ISA 16-bit A/B layouts + 2x2 WMMA ----
        Frag16 af[2], bf[2];
#pragma unroll
        for (int j = 0; j < 8; ++j) {
            int kA = (j < 4) ? (half * 8 + 2 * j) : (16 + half * 8 + 2 * (j - 4));
            af[0].w[j] = *reinterpret_cast<const unsigned int*>(&As[(wm + lm) * BK + kA]);
            af[1].w[j] = *reinterpret_cast<const unsigned int*>(&As[(wm + 16 + lm) * BK + kA]);
            int kB = (j < 4) ? (half * 8 + 2 * j) : (16 + half * 8 + 2 * (j - 4));
            bf[0].w[j] = *reinterpret_cast<const unsigned int*>(&Bs[(wn + lm) * BK + kB]);
            bf[1].w[j] = *reinterpret_cast<const unsigned int*>(&Bs[(wn + 16 + lm) * BK + kB]);
        }
#pragma unroll
        for (int mi = 0; mi < 2; ++mi)
#pragma unroll
            for (int ni = 0; ni < 2; ++ni)
                acc[mi][ni] = __builtin_amdgcn_wmma_f32_16x16x32_bf16(
                    false, af[mi].v, false, bf[ni].v, (short)0, acc[mi][ni],
                    false, false);
        __syncthreads();

        // ---- commit next K-step into LDS ----
        if (has_next) {
            *reinterpret_cast<uint4*>(&As[am * BK + ak]) = av;
#pragma unroll
            for (int i = 0; i < 8; ++i)
                *reinterpret_cast<unsigned int*>(&Bs[bn * BK + (bkp0 + i * 2) * 2]) =
                    pack2bf(w0r[i], w1r[i]);
        }
        __syncthreads();
    }

    // ---- epilogue: C layout M = half*8 + r, N = lm ----
#pragma unroll
    for (int mi = 0; mi < 2; ++mi)
#pragma unroll
        for (int ni = 0; ni < 2; ++ni)
#pragma unroll
            for (int r = 0; r < 8; ++r) {
                int M = m0 + wm + mi * 16 + half * 8 + r;
                int N = n0 + wn + ni * 16 + lm;
                size_t oi = (size_t)(M * 16 + g) * DOUT + N;
                if (OUT_F32)
                    reinterpret_cast<float*>(Out)[oi] = acc[mi][ni][r];
                else
                    reinterpret_cast<unsigned short*>(Out)[oi] = f2bf(acc[mi][ni][r]);
            }
}

// ---------------------------------------------------------------------------
// Attention: one block per batch b, one wave per head h (8 waves).
// scores = q k^T / 8  (two chained WMMAs over K=64), softmax across 16 keys,
// row mask, then ctx = probs x v via 32 WMMAs (K padded 16->32 with zeros).
// ---------------------------------------------------------------------------
__global__ __launch_bounds__(256) void attention_kernel(
    const unsigned short* __restrict__ q,    // (256,16,512) bf16
    const unsigned short* __restrict__ k_,   // (256,16,512) bf16
    const unsigned short* __restrict__ v,    // (256,16,4096) bf16
    const int* __restrict__ mask,            // (256,16)
    unsigned short* __restrict__ ctx)        // (256,16,4096) bf16
{
    __shared__ __align__(16) unsigned short P[8][16 * 32];  // per-wave probs, K-padded

    const int b    = blockIdx.x;
    const int t    = threadIdx.x;
    const int h    = t >> 5;      // head = wave id
    const int lane = t & 31;
    const int lm   = lane & 15;
    const int half = lane >> 4;

    // ---- scores: A = q (16x64), B = k^T (64x16); fragments direct from global
    v8f c = {};
#pragma unroll
    for (int s = 0; s < 2; ++s) {
        Frag16 a, bb;
#pragma unroll
        for (int j = 0; j < 8; ++j) {
            int kA = s * 32 + ((j < 4) ? (half * 8 + 2 * j)
                                       : (16 + half * 8 + 2 * (j - 4)));
            a.w[j] = *reinterpret_cast<const unsigned int*>(
                &q[(size_t)(b * 16 + lm) * 512 + h * 64 + kA]);
            int kB = s * 32 + ((j < 4) ? (half * 8 + 2 * j)
                                       : (16 + half * 8 + 2 * (j - 4)));
            bb.w[j] = *reinterpret_cast<const unsigned int*>(
                &k_[(size_t)(b * 16 + lm) * 512 + h * 64 + kB]);
        }
        c = __builtin_amdgcn_wmma_f32_16x16x32_bf16(false, a.v, false, bb.v,
                                                    (short)0, c, false, false);
    }

    // ---- softmax along keys (16 lanes of each half hold one row) + row mask
    float pr[8];
#pragma unroll
    for (int r = 0; r < 8; ++r) {
        float s = c[r] * 0.125f;   // 1/sqrt(64)
        float mx = s;
        mx = fmaxf(mx, swz_xor<8>(mx));
        mx = fmaxf(mx, swz_xor<4>(mx));
        mx = fmaxf(mx, swz_xor<2>(mx));
        mx = fmaxf(mx, swz_xor<1>(mx));
        float e = __expf(s - mx);
        float sum = e;
        sum += swz_xor<8>(sum);
        sum += swz_xor<4>(sum);
        sum += swz_xor<2>(sum);
        sum += swz_xor<1>(sum);
        float p = e / sum;
        int row = half * 8 + r;                       // query index
        p *= (float)mask[b * 16 + row];               // post-softmax row mask
        pr[r] = p;
    }

    // ---- transpose probs through LDS into A-fragment layout (zero K pad) ----
#pragma unroll
    for (int r = 0; r < 8; ++r) {
        int row = half * 8 + r;
        P[h][row * 32 + lm] = f2bf(pr[r]);            // real K = 0..15
        P[h][lm * 32 + 16 + half * 8 + r] = 0;        // pad K = 16..31
    }
    __syncthreads();

    Frag16 ap;
#pragma unroll
    for (int j = 0; j < 8; ++j) {
        int kA = half * 16 + 2 * j;
        ap.w[j] = *reinterpret_cast<const unsigned int*>(&P[h][lm * 32 + kA]);
    }

    // ---- ctx = probs (16x16, K-padded) x v (16x512): 32 N-tiles of 16 ----
    for (int nt = 0; nt < 32; ++nt) {
        Frag16 bv;
#pragma unroll
        for (int j = 0; j < 8; ++j) {
            unsigned int val = 0;
            if (half == 0) {                          // K = 16..31 rows are zero
                int kk = 2 * j;
                unsigned short v0 = v[(size_t)(b * 16 + kk) * 4096 + h * 512 + nt * 16 + lm];
                unsigned short v1 = v[(size_t)(b * 16 + kk + 1) * 4096 + h * 512 + nt * 16 + lm];
                val = (unsigned int)v0 | ((unsigned int)v1 << 16);
            }
            bv.w[j] = val;
        }
        v8f cc = {};
        cc = __builtin_amdgcn_wmma_f32_16x16x32_bf16(false, ap.v, false, bv.v,
                                                     (short)0, cc, false, false);
#pragma unroll
        for (int r = 0; r < 8; ++r) {
            int nq = half * 8 + r;
            ctx[(size_t)(b * 16 + nq) * 4096 + h * 512 + nt * 16 + lm] = f2bf(cc[r]);
        }
    }
}

// ---------------------------------------------------------------------------
// Host-side launcher
// ---------------------------------------------------------------------------
extern "C" void kernel_launch(void* const* d_in, const int* in_sizes, int n_in,
                              void* d_out, int out_size, void* d_ws, size_t ws_size,
                              hipStream_t stream) {
    (void)in_sizes; (void)n_in; (void)out_size; (void)ws_size;

    const float* h    = (const float*)d_in[0];   // (256,16,512)
    const int*   mask = (const int*)  d_in[1];   // (256,16)
    const float* Wk   = (const float*)d_in[2];   // (16,512,512)
    const float* Wq   = (const float*)d_in[3];   // (16,512,512)
    const float* Wv   = (const float*)d_in[4];   // (16,512,4096)
    const float* Wo   = (const float*)d_in[5];   // (16,4096,512)
    float* out = (float*)d_out;                  // (256,16,512)

    // workspace layout (bf16 intermediates, MiB-aligned offsets)
    char* ws = (char*)d_ws;
    unsigned short* h_bf = (unsigned short*)(ws + 0);                    //  4 MiB
    unsigned short* qb   = (unsigned short*)(ws + (4u  << 20));          //  4 MiB
    unsigned short* kb   = (unsigned short*)(ws + (8u  << 20));          //  4 MiB
    unsigned short* vb   = (unsigned short*)(ws + (12u << 20));          // 32 MiB
    unsigned short* ctxb = (unsigned short*)(ws + (44u << 20));          // 32 MiB

    const int nH = 256 * 16 * 512;
    cvt_f32_bf16<<<(nH + 255) / 256, 256, 0, stream>>>(h, h_bf, nH);

    // grouped QKV projections (bf16 out)
    grouped_gemm_bf16<512, 512,  false><<<dim3(4,  4, 16), 256, 0, stream>>>(h_bf, Wk, kb);
    grouped_gemm_bf16<512, 512,  false><<<dim3(4,  4, 16), 256, 0, stream>>>(h_bf, Wq, qb);
    grouped_gemm_bf16<512, 4096, false><<<dim3(32, 4, 16), 256, 0, stream>>>(h_bf, Wv, vb);

    // per-head attention (scores, softmax, mask, probs x v)
    attention_kernel<<<dim3(256), 256, 0, stream>>>(qb, kb, vb, mask, ctxb);

    // output projection (f32 out, straight to d_out)
    grouped_gemm_bf16<4096, 512, true><<<dim3(4, 4, 16), 256, 0, stream>>>(ctxb, Wo, out);
}
